// CausalMultiheadSelfAttention_68135361184473
// MI455X (gfx1250) — matrix operand order repaired and checked by reading, not compiled, not run on hardware
//
#include <hip/hip_runtime.h>
#include <hip/hip_bf16.h>

typedef unsigned short u16;
typedef unsigned int   u32;
typedef unsigned long long u64;
typedef __attribute__((ext_vector_type(8)))  float  v8f;
typedef __attribute__((ext_vector_type(16))) __bf16 v16bf;
typedef __attribute__((ext_vector_type(8)))  __bf16 v8bf;

constexpr int D_MODEL   = 1024;
constexpr int NUM_HEADS = 16;
constexpr int D_K       = 64;
constexpr int BATCH     = 4;
constexpr int SEQ       = 2048;
constexpr int MTOT      = BATCH * SEQ;   // 8192 rows for the big GEMMs

// ---------- helpers ----------

__device__ __forceinline__ u16 f2bf(float f) {
    u32 u = __float_as_uint(f);
    return (u16)((u + 0x7FFFu + ((u >> 16) & 1u)) >> 16);
}

__device__ __forceinline__ v16bf frag16(const u16* lo, const u16* hi) {
    v8bf a = *(const v8bf*)lo;
    v8bf b = *(const v8bf*)hi;
    return __builtin_shufflevector(a, b, 0,1,2,3,4,5,6,7,8,9,10,11,12,13,14,15);
}

// gfx1250 async copy global->LDS (ASYNCcnt-tracked, bypasses VGPRs).
// lds = workgroup LDS byte offset (low 32 bits of a generic __shared__ pointer),
// g   = 64-bit global address. GV addressing mode.
__device__ __forceinline__ void async_cp_b128(u32 lds, const void* g) {
    asm volatile("global_load_async_to_lds_b128 %0, %1, off"
                 :: "v"(lds), "v"((u64)(size_t)g) : "memory");
}

__device__ __forceinline__ void wait_async0() {
    asm volatile("s_wait_asynccnt 0x0" ::: "memory");
}

__device__ __forceinline__ u32 lds_off(const void* p) {
    return (u32)(size_t)p;   // generic LDS pointer: low 32 bits = LDS byte offset
}

// ---------- f32 -> bf16 conversion (vectorized, grid-stride) ----------

__global__ void cvt_f32_bf16_k(const float* __restrict__ src, u16* __restrict__ dst, int n4) {
    for (int i = blockIdx.x * blockDim.x + threadIdx.x; i < n4;
         i += gridDim.x * blockDim.x) {
        float4 v = ((const float4*)src)[i];
        uint2 o;
        o.x = (u32)f2bf(v.x) | ((u32)f2bf(v.y) << 16);
        o.y = (u32)f2bf(v.z) | ((u32)f2bf(v.w) << 16);
        ((uint2*)dst)[i] = o;
    }
}

// ---------- GEMM: C[m,n] = sum_k A[m,k] * W[n,k] + bias[n] ----------
// A: (MTOT x 1024) bf16 row-major.  W: (1024 x 1024) bf16 row-major (row n = output n).
// MODE 0: write bf16 into (B,H,S,d_k) layout (QKV).  MODE 1: write f32 (B*S,1024).
// Block = 256 threads (8 waves), tile 128x128, K-step 32, wave = 64x32 sub-tile.
// Double-buffered LDS staging via global_load_async_to_lds_b128.

template <int MODE>
__global__ __launch_bounds__(256) void gemm_bf16_k(const u16* __restrict__ A,
                                                   const u16* __restrict__ W,
                                                   const float* __restrict__ bias,
                                                   void* __restrict__ outp) {
    __shared__ u16 As[2][128 * 40];   // +8 halves pad per row -> conflict-free b128 reads
    __shared__ u16 Bs[2][128 * 40];

    const int tid   = threadIdx.x;
    const int lane  = tid & 31;
    const int wave  = tid >> 5;
    const int lhalf = lane & 15;
    const int hi    = lane >> 4;
    const int wm    = wave >> 2;   // 0..1  (64 rows each)
    const int wn    = wave & 3;    // 0..3  (32 cols each)
    const int m0    = blockIdx.y * 128;
    const int n0    = blockIdx.x * 128;

    // Per-thread staging chunks (2 for A, 2 for B), fixed (row, col) per thread.
    int    rowc[2], ccc[2];
    u32    ldsA[2][2], ldsB[2][2];          // [buf][chunk]
    size_t goffA[2], goffB[2];              // global element offsets (excl. k0)
#pragma unroll
    for (int i = 0; i < 2; ++i) {
        const int cid = tid + i * 256;
        rowc[i] = cid >> 2;
        ccc[i]  = (cid & 3) * 8;
        goffA[i] = (size_t)(m0 + rowc[i]) * D_MODEL + ccc[i];
        goffB[i] = (size_t)(n0 + rowc[i]) * D_MODEL + ccc[i];
#pragma unroll
        for (int b = 0; b < 2; ++b) {
            ldsA[b][i] = lds_off(&As[b][rowc[i] * 40 + ccc[i]]);
            ldsB[b][i] = lds_off(&Bs[b][rowc[i] * 40 + ccc[i]]);
        }
    }

    v8f acc[4][2];
#pragma unroll
    for (int mi = 0; mi < 4; ++mi)
#pragma unroll
        for (int ni = 0; ni < 2; ++ni)
#pragma unroll
            for (int j = 0; j < 8; ++j) acc[mi][ni][j] = 0.0f;

    // Prologue: prefetch tile 0 into buffer 0.
#pragma unroll
    for (int i = 0; i < 2; ++i) {
        async_cp_b128(ldsA[0][i], A + goffA[i]);
        async_cp_b128(ldsB[0][i], W + goffB[i]);
    }

    for (int k0 = 0; k0 < D_MODEL; k0 += 32) {
        const int buf = (k0 >> 5) & 1;
        wait_async0();        // my async copies for tile k0 have landed in LDS
        __syncthreads();      // everyone's copies landed; prev compute done reading buf^1

        if (k0 + 32 < D_MODEL) {   // prefetch next tile into the other buffer
#pragma unroll
            for (int i = 0; i < 2; ++i) {
                async_cp_b128(ldsA[buf ^ 1][i], A + goffA[i] + k0 + 32);
                async_cp_b128(ldsB[buf ^ 1][i], W + goffB[i] + k0 + 32);
            }
        }

        const u16* as = &As[buf][0];
        const u16* bs = &Bs[buf][0];
        v16bf af[4], bf_[2];
#pragma unroll
        for (int mi = 0; mi < 4; ++mi) {
            int r = wm * 64 + mi * 16 + lhalf;
            af[mi] = frag16(&as[r * 40 + hi * 8], &as[r * 40 + hi * 8 + 16]);
        }
#pragma unroll
        for (int ni = 0; ni < 2; ++ni) {
            int r = wn * 32 + ni * 16 + lhalf;
            bf_[ni] = frag16(&bs[r * 40 + hi * 8], &bs[r * 40 + hi * 8 + 16]);
        }
#pragma unroll
        for (int mi = 0; mi < 4; ++mi)
#pragma unroll
            for (int ni = 0; ni < 2; ++ni)
                acc[mi][ni] = __builtin_amdgcn_wmma_f32_16x16x32_bf16(
                    false, af[mi], false, bf_[ni], (short)0, acc[mi][ni], false, false);
    }

#pragma unroll
    for (int ni = 0; ni < 2; ++ni) {
        const int n  = n0 + wn * 32 + ni * 16 + lhalf;
        const float bv = bias[n];
#pragma unroll
        for (int mi = 0; mi < 4; ++mi) {
#pragma unroll
            for (int j = 0; j < 8; ++j) {
                const int m = m0 + wm * 64 + mi * 16 + j + 8 * hi;
                const float val = acc[mi][ni][j] + bv;
                if (MODE == 0) {
                    const int bb = m >> 11, ss = m & 2047;
                    const int hh = n >> 6,  dd = n & 63;
                    ((u16*)outp)[((size_t)(bb * NUM_HEADS + hh) * SEQ + ss) * D_K + dd] =
                        f2bf(val);
                } else {
                    ((float*)outp)[(size_t)m * D_MODEL + n] = val;
                }
            }
        }
    }
}

// ---------- Flash attention (causal, online softmax) ----------
// Grid: (SEQ/64, BATCH*NUM_HEADS). Block 128 threads = 4 waves; wave w owns q rows
// [q0+16w, q0+16w+15]. Streams 64-key tiles; never materializes the SxS matrix.
// K tile staged with async-to-LDS, overlapped with the in-VGPR V transpose.

__global__ __launch_bounds__(128) void flash_attn_k(const u16* __restrict__ Q,
                                                    const u16* __restrict__ K,
                                                    const u16* __restrict__ V,
                                                    u16* __restrict__ O) {
    __shared__ u16 Qs[64 * 72];       // q rows, row-major (+8 pad)
    __shared__ u16 Ks[64 * 72];       // key rows, row-major
    __shared__ u16 Vt[64 * 72];       // transposed: [d_k][kv]
    __shared__ u16 Ps[4 * 16 * 72];   // per-wave P staging (C-layout -> A-layout)

    const int tid   = threadIdx.x;
    const int lane  = tid & 31;
    const int w     = tid >> 5;
    const int lhalf = lane & 15;
    const int hi    = lane >> 4;
    const int q0    = blockIdx.x * 64;
    const int bh    = blockIdx.y;
    const int bb    = bh >> 4;
    const int hh    = bh & 15;

    const u16* Qh = Q + (size_t)bh * SEQ * D_K;
    const u16* Kh = K + (size_t)bh * SEQ * D_K;
    const u16* Vh = V + (size_t)bh * SEQ * D_K;

    int rowv[4], ccv[4];
    u32 ldsK[4];
#pragma unroll
    for (int i = 0; i < 4; ++i) {
        const int cid = tid + i * 128;
        rowv[i] = cid >> 3;
        ccv[i]  = (cid & 7) * 8;
        ldsK[i] = lds_off(&Ks[rowv[i] * 72 + ccv[i]]);
    }

    // Load Q tile (64 x 64 bf16) once.
#pragma unroll
    for (int i = 0; i < 4; ++i) {
        *(uint4*)&Qs[rowv[i] * 72 + ccv[i]] =
            *(const uint4*)&Qh[(size_t)(q0 + rowv[i]) * D_K + ccv[i]];
    }
    __syncthreads();

    v16bf qa[2];
#pragma unroll
    for (int kc = 0; kc < 2; ++kc) {
        int r = w * 16 + lhalf;
        qa[kc] = frag16(&Qs[r * 72 + kc * 32 + hi * 8],
                        &Qs[r * 72 + kc * 32 + 16 + hi * 8]);
    }

    float mrow[8], lrow[8];
    v8f o[4];
#pragma unroll
    for (int j = 0; j < 8; ++j) { mrow[j] = -__builtin_inff(); lrow[j] = 0.0f; }
#pragma unroll
    for (int nd = 0; nd < 4; ++nd)
#pragma unroll
        for (int j = 0; j < 8; ++j) o[nd][j] = 0.0f;

    const float smul = 0.125f;   // 1/sqrt(64)

    for (int kv0 = 0; kv0 < q0 + 64; kv0 += 64) {
        __syncthreads();   // all waves done reading Ks/Vt from previous tile

        // Async-stage K tile (hidden behind the V transpose below).
#pragma unroll
        for (int i = 0; i < 4; ++i)
            async_cp_b128(ldsK[i], &Kh[(size_t)(kv0 + rowv[i]) * D_K + ccv[i]]);

        // Transpose-stage V tile through VGPRs.
#pragma unroll
        for (int i = 0; i < 4; ++i) {
            union { uint4 q; u16 s[8]; } t;
            t.q = *(const uint4*)&Vh[(size_t)(kv0 + rowv[i]) * D_K + ccv[i]];
#pragma unroll
            for (int j = 0; j < 8; ++j) Vt[(ccv[i] + j) * 72 + rowv[i]] = t.s[j];
        }
        wait_async0();
        __syncthreads();

        // S = Q K^T  (wave: 16 x 64 scores = 4 accumulators)
        v8f sc[4];
#pragma unroll
        for (int ni = 0; ni < 4; ++ni) {
#pragma unroll
            for (int j = 0; j < 8; ++j) sc[ni][j] = 0.0f;
#pragma unroll
            for (int kc = 0; kc < 2; ++kc) {
                int r = ni * 16 + lhalf;
                v16bf bfrag = frag16(&Ks[r * 72 + kc * 32 + hi * 8],
                                     &Ks[r * 72 + kc * 32 + 16 + hi * 8]);
                sc[ni] = __builtin_amdgcn_wmma_f32_16x16x32_bf16(
                    false, qa[kc], false, bfrag, (short)0, sc[ni], false, false);
            }
        }

        // Causal mask + row max (C-layout: VGPR j -> row j + 8*hi, lane -> column).
        float rmax[8];
#pragma unroll
        for (int j = 0; j < 8; ++j) rmax[j] = -__builtin_inff();
#pragma unroll
        for (int ni = 0; ni < 4; ++ni) {
            const int kvc = kv0 + ni * 16 + lhalf;
#pragma unroll
            for (int j = 0; j < 8; ++j) {
                const int qr = q0 + w * 16 + j + 8 * hi;
                float x = sc[ni][j] * smul;
                x = (kvc > qr) ? -__builtin_inff() : x;
                sc[ni][j] = x;
                rmax[j] = fmaxf(rmax[j], x);
            }
        }
#pragma unroll
        for (int j = 0; j < 8; ++j)
#pragma unroll
            for (int off = 1; off < 16; off <<= 1)
                rmax[j] = fmaxf(rmax[j], __shfl_xor(rmax[j], off, 16));

        float scf[8], rs[8];
#pragma unroll
        for (int j = 0; j < 8; ++j) {
            const float mn = fmaxf(mrow[j], rmax[j]);
            scf[j]  = __expf(mrow[j] - mn);
            mrow[j] = mn;
            lrow[j] *= scf[j];
            rs[j] = 0.0f;
        }
#pragma unroll
        for (int ni = 0; ni < 4; ++ni)
#pragma unroll
            for (int j = 0; j < 8; ++j) {
                const float p = __expf(sc[ni][j] - mrow[j]);
                sc[ni][j] = p;
                rs[j] += p;
            }
#pragma unroll
        for (int j = 0; j < 8; ++j) {
#pragma unroll
            for (int off = 1; off < 16; off <<= 1)
                rs[j] += __shfl_xor(rs[j], off, 16);
            lrow[j] += rs[j];
        }
#pragma unroll
        for (int nd = 0; nd < 4; ++nd)
#pragma unroll
            for (int j = 0; j < 8; ++j) o[nd][j] *= scf[j];

        // Stage P (bf16) through LDS to convert C-layout -> A-layout.
        u16* pb = &Ps[w * 16 * 72];
#pragma unroll
        for (int ni = 0; ni < 4; ++ni)
#pragma unroll
            for (int j = 0; j < 8; ++j)
                pb[(j + 8 * hi) * 72 + ni * 16 + lhalf] = f2bf(sc[ni][j]);
        __syncthreads();

        v16bf ap[2];
#pragma unroll
        for (int kc = 0; kc < 2; ++kc)
            ap[kc] = frag16(&pb[lhalf * 72 + kc * 32 + hi * 8],
                            &pb[lhalf * 72 + kc * 32 + 16 + hi * 8]);

        // O += P V  (B fragment = contiguous rows of Vt)
#pragma unroll
        for (int nd = 0; nd < 4; ++nd)
#pragma unroll
            for (int kc = 0; kc < 2; ++kc) {
                int r = nd * 16 + lhalf;
                v16bf bv = frag16(&Vt[r * 72 + kc * 32 + hi * 8],
                                  &Vt[r * 72 + kc * 32 + 16 + hi * 8]);
                o[nd] = __builtin_amdgcn_wmma_f32_16x16x32_bf16(
                    false, ap[kc], false, bv, (short)0, o[nd], false, false);
            }
    }

    // Finalize: divide by row sum, write (B,S,H,d_k) bf16 (ready for out-proj GEMM).
    float li[8];
#pragma unroll
    for (int j = 0; j < 8; ++j) li[j] = 1.0f / lrow[j];
#pragma unroll
    for (int nd = 0; nd < 4; ++nd) {
        const int dd = nd * 16 + lhalf;
#pragma unroll
        for (int j = 0; j < 8; ++j) {
            const int qr = q0 + w * 16 + j + 8 * hi;
            O[((size_t)(bb * SEQ + qr) * NUM_HEADS + hh) * D_K + dd] =
                f2bf(o[nd][j] * li[j]);
        }
    }
}

// ---------- launch ----------

extern "C" void kernel_launch(void* const* d_in, const int* in_sizes, int n_in,
                              void* d_out, int out_size, void* d_ws, size_t ws_size,
                              hipStream_t stream) {
    const float* x  = (const float*)d_in[0];
    const float* Wq = (const float*)d_in[1];
    const float* bq = (const float*)d_in[2];
    const float* Wk = (const float*)d_in[3];
    const float* bk = (const float*)d_in[4];
    const float* Wv = (const float*)d_in[5];
    const float* bv = (const float*)d_in[6];
    const float* Wo = (const float*)d_in[7];
    const float* bo = (const float*)d_in[8];

    const size_t XN = (size_t)MTOT * D_MODEL;      // 8,388,608
    const size_t WN = (size_t)D_MODEL * D_MODEL;   // 1,048,576

    u16* ws  = (u16*)d_ws;
    u16* xb  = ws;
    u16* wqb = xb  + XN;
    u16* wkb = wqb + WN;
    u16* wvb = wkb + WN;
    u16* wob = wvb + WN;
    u16* Qb  = wob + WN;
    u16* Kb  = Qb  + XN;
    u16* Vb  = Kb  + XN;
    u16* Ob  = Vb  + XN;   // total 92,274,688 bytes

    // 1) bf16 conversions
    {
        int n4 = (int)(XN / 4);
        cvt_f32_bf16_k<<<dim3((n4 + 255) / 256), dim3(256), 0, stream>>>(x, xb, n4);
        int w4 = (int)(WN / 4);
        cvt_f32_bf16_k<<<dim3((w4 + 255) / 256), dim3(256), 0, stream>>>(Wq, wqb, w4);
        cvt_f32_bf16_k<<<dim3((w4 + 255) / 256), dim3(256), 0, stream>>>(Wk, wkb, w4);
        cvt_f32_bf16_k<<<dim3((w4 + 255) / 256), dim3(256), 0, stream>>>(Wv, wvb, w4);
        cvt_f32_bf16_k<<<dim3((w4 + 255) / 256), dim3(256), 0, stream>>>(Wo, wob, w4);
    }

    // 2) Q/K/V projections (write (B,H,S,d_k) bf16)
    const dim3 ggrid(D_MODEL / 128, MTOT / 128);   // (8, 64)
    gemm_bf16_k<0><<<ggrid, dim3(256), 0, stream>>>(xb, wqb, bq, Qb);
    gemm_bf16_k<0><<<ggrid, dim3(256), 0, stream>>>(xb, wkb, bk, Kb);
    gemm_bf16_k<0><<<ggrid, dim3(256), 0, stream>>>(xb, wvb, bv, Vb);

    // 3) Causal flash attention -> (B,S,H*d_k) bf16
    flash_attn_k<<<dim3(SEQ / 64, BATCH * NUM_HEADS), dim3(128), 0, stream>>>(Qb, Kb, Vb, Ob);

    // 4) Output projection -> f32 d_out (B,S,D)
    gemm_bf16_k<1><<<ggrid, dim3(256), 0, stream>>>(Ob, wob, bo, (float*)d_out);
}
